// Net_23287312678947
// MI455X (gfx1250) — matrix operand order repaired and checked by reading, not compiled, not run on hardware
//
#include <hip/hip_runtime.h>

typedef __bf16 bf16;
typedef __attribute__((ext_vector_type(16))) __bf16 v16bf;
typedef __attribute__((ext_vector_type(8)))  float  v8f;

#define BGR  32
#define NPG  1024
#define KNN  24
#define HDIM 64
#define NTOT (BGR*NPG)
#define EPSV 1e-5f

__device__ __forceinline__ float eluf(float x) { return x > 0.f ? x : __expf(x) - 1.f; }

// ---- WMMA fragment loaders (wave32), contiguous k = 16*(lane/16) + e for A and B ----
// A: 16x32 bf16. lane<16: K=0..15 ; lane>=16: K=16..31 (contiguous, matches B loaders)
__device__ __forceinline__ v16bf load_a_frag(const bf16* s, int ld, int lane) {
  v16bf a; int m = lane & 15; int kh = (lane >> 4) << 4;
#pragma unroll
  for (int e = 0; e < 16; ++e) {
    int v = e >> 1, p = e & 1;
    int k = kh + 2 * v + p;
    a[e] = s[m * ld + k];
  }
  return a;
}
// B: 32x16 bf16 from row-major [K][N]: lanes 0-15 K=0..15, lanes 16-31 K=16..31
__device__ __forceinline__ v16bf load_b_frag(const bf16* s, int ld, int lane) {
  v16bf b; int n = lane & 15; int kb = (lane >> 4) << 4;
#pragma unroll
  for (int e = 0; e < 16; ++e) b[e] = s[(kb + e) * ld + n];
  return b;
}
// B: 32x16 where element(k,n) = src[n*ld + k]  (B = X^T from row-major X)
__device__ __forceinline__ v16bf load_bT_frag(const bf16* s, int ld, int lane) {
  v16bf b; int n = lane & 15; int kb = (lane >> 4) << 4;
#pragma unroll
  for (int e = 0; e < 16; ++e) b[e] = s[n * ld + kb + e];
  return b;
}
__device__ __forceinline__ v8f wmma_bf(v16bf a, v16bf b, v8f c) {
  return __builtin_amdgcn_wmma_f32_16x16x32_bf16(false, a, false, b, (short)0, c, false, false);
}
__device__ __forceinline__ void atomicMaxF(float* p, float v) {
  unsigned* u = (unsigned*)p;
  unsigned old = *u;
  while (__uint_as_float(old) < v) {
    unsigned assumed = old;
    old = atomicCAS(u, assumed, __float_as_uint(v));
    if (old == assumed) break;
  }
}

// ---- weight fp32 -> bf16 (conv w1:128x64, w2:64x64, 3 layers) ----
__global__ void cvt_w_kernel(const float* w1a, const float* w2a, const float* w1b,
                             const float* w2b, const float* w1c, const float* w2c,
                             bf16* wbf) {
  int i = blockIdx.x * blockDim.x + threadIdx.x;
  if (i >= 3 * 12288) return;
  int l = i / 12288, rem = i % 12288;
  const float* w1 = (l == 0) ? w1a : (l == 1) ? w1b : w1c;
  const float* w2 = (l == 0) ? w2a : (l == 1) ? w2b : w2c;
  float v = (rem < 8192) ? w1[rem] : w2[rem - 8192];
  wbf[i] = (bf16)v;
}

// ---- encoder MLP: [N,8] -> [N,64], 32 nodes/block, 256 threads ----
__global__ void enc_kernel(const float* __restrict__ xin,
                           const float* w1, const float* b1, const float* w2,
                           const float* b2, const float* w3, const float* b3,
                           float* __restrict__ xf) {
  __shared__ float sw1[8 * 64], sb1[64], sw2[64 * 64], sb2[64], sw3[64 * 64], sb3[64];
  __shared__ float sin0[32 * 8], hA[32 * 64], hB[32 * 64];
  int tid = threadIdx.x;
  for (int i = tid; i < 8 * 64; i += 256) sw1[i] = w1[i];
  for (int i = tid; i < 64 * 64; i += 256) { sw2[i] = w2[i]; sw3[i] = w3[i]; }
  if (tid < 64) { sb1[tid] = b1[tid]; sb2[tid] = b2[tid]; sb3[tid] = b3[tid]; }
  int n0 = blockIdx.x * 32;
  for (int i = tid; i < 32 * 8; i += 256) sin0[i] = xin[n0 * 8 + i];
  __syncthreads();
  int ln = tid >> 3, g = tid & 7;
#pragma unroll
  for (int q = 0; q < 8; ++q) {
    int c = g * 8 + q; float a = sb1[c];
#pragma unroll
    for (int k = 0; k < 8; ++k) a += sin0[ln * 8 + k] * sw1[k * 64 + c];
    hA[ln * 64 + c] = eluf(a);
  }
  __syncthreads();
#pragma unroll
  for (int q = 0; q < 8; ++q) {
    int c = g * 8 + q; float a = sb2[c];
    for (int k = 0; k < 64; ++k) a += hA[ln * 64 + k] * sw2[k * 64 + c];
    hB[ln * 64 + c] = eluf(a);
  }
  __syncthreads();
#pragma unroll
  for (int q = 0; q < 8; ++q) {
    int c = g * 8 + q; float a = sb3[c];
    for (int k = 0; k < 64; ++k) a += hB[ln * 64 + k] * sw3[k * 64 + c];
    xf[(n0 + ln) * 64 + c] = a;
  }
}

// ---- bf16 mirror + squared norms for kNN ----
__global__ void prep_kernel(const float* __restrict__ xf, bf16* __restrict__ xbf,
                            float* __restrict__ sq) {
  int n = blockIdx.x * blockDim.x + threadIdx.x;
  if (n >= NTOT) return;
  float s = 0.f;
#pragma unroll
  for (int c = 0; c < 64; ++c) {
    float v = xf[n * 64 + c];
    s += v * v;
    xbf[n * 64 + c] = (bf16)v;
  }
  sq[n] = s;
}

// ---- kNN: WMMA Gram tiles + incremental top-24 (16 rows/block, 4 waves) ----
__global__ void knn_kernel(const bf16* __restrict__ xbf, const float* __restrict__ sq,
                           int* __restrict__ idx) {
  __shared__ bf16 arow[16 * 64];
  __shared__ float sqrow[16];
  __shared__ float d2c[16 * 64];
  int tid = threadIdx.x;
  int b = blockIdx.x >> 6;
  int r0 = (blockIdx.x & 63) << 4;
  const bf16* xg = xbf + (size_t)b * NPG * HDIM;
  const float* sqg = sq + b * NPG;
  for (int i = tid; i < 16 * 64; i += 128) arow[i] = xg[(r0 + (i >> 6)) * HDIM + (i & 63)];
  if (tid < 16) sqrow[tid] = sqg[r0 + tid];
  __syncthreads();
  int lane = tid & 31, wv = tid >> 5;
  // A fragments are invariant across all column chunks: hoist into registers.
  v16bf afr[2];
#pragma unroll
  for (int kb = 0; kb < 2; ++kb) afr[kb] = load_a_frag(arow + kb * 32, HDIM, lane);
  float vals[KNN]; int ids[KNN]; float maxv = 3.0e38f; int maxp = 0;
#pragma unroll
  for (int q = 0; q < KNN; ++q) { vals[q] = 3.0e38f; ids[q] = 0; }
  for (int chunk = 0; chunk < 16; ++chunk) {
    int c0 = (chunk << 6) + (wv << 4);
    v8f acc = {};
#pragma unroll
    for (int kb = 0; kb < 2; ++kb) {
      v16bf bf = load_bT_frag(xg + (size_t)c0 * HDIM + kb * 32, HDIM, lane);
      acc = wmma_bf(afr[kb], bf, acc);
    }
    float sqn = sqg[c0 + (lane & 15)];
#pragma unroll
    for (int r = 0; r < 8; ++r) {
      int m = r + ((lane >> 4) << 3);
      int n = lane & 15;
      d2c[m * 64 + (wv << 4) + n] = sqrow[m] + sqn - 2.0f * acc[r];
    }
    __syncthreads();
    if (tid < 16) {
      for (int j = 0; j < 64; ++j) {
        float v = d2c[tid * 64 + j];
        if (v < maxv) {                 // replace current max, rescan
          vals[maxp] = v; ids[maxp] = (chunk << 6) + j;
          maxv = vals[0]; maxp = 0;
#pragma unroll
          for (int q = 1; q < KNN; ++q)
            if (vals[q] > maxv) { maxv = vals[q]; maxp = q; }
        }
      }
    }
    __syncthreads();
  }
  if (tid < 16) {
    int gn = b * NPG + r0 + tid;
#pragma unroll
    for (int q = 0; q < KNN; ++q) idx[gn * KNN + q] = ids[q];
  }
}

// ---- edge MLP (fused gather + GEMM1 + GEMM2 + max-agg), 4 nodes/block ----
__global__ void edge_kernel(const float* __restrict__ xf, const bf16* __restrict__ wbf,
                            const float* __restrict__ b1, const float* __restrict__ b2,
                            const int* __restrict__ idx, float* __restrict__ feats) {
  __shared__ bf16 eL[96 * 128];     // [xi, xj-xi] rows
  __shared__ bf16 h1L[96 * 64];
  __shared__ bf16 w1L[128 * 64];
  __shared__ bf16 w2L[64 * 64];
  __shared__ float b1L[64], b2L[64];
  __shared__ float outL[4 * 64];
  __shared__ int idxL[96];
  int tid = threadIdx.x;
  int node0 = blockIdx.x << 2;
  for (int i = tid; i < 128 * 64; i += 128) w1L[i] = wbf[i];
  for (int i = tid; i < 64 * 64; i += 128) w2L[i] = wbf[8192 + i];
  if (tid < 64) { b1L[tid] = b1[tid]; b2L[tid] = b2[tid]; }
  for (int i = tid; i < 256; i += 128) outL[i] = -3.0e38f;
  if (tid < 96) {
    int nd = tid / KNN, kk = tid % KNN;
    int j = idx[(node0 + nd) * KNN + kk];
    idxL[tid] = j;
    int gb = ((node0 + nd) / NPG) * NPG;
    __builtin_prefetch(&xf[(size_t)(gb + j) * HDIM], 0, 1);  // global_prefetch_b8
  }
  __syncthreads();
  for (int i = tid; i < 96 * 128; i += 128) {
    int r = i >> 7, c = i & 127;
    int nd = r / KNN;
    int gi = node0 + nd;
    float v;
    if (c < 64) {
      v = xf[gi * HDIM + c];
    } else {
      int gb = (gi / NPG) * NPG;
      int j = idxL[r];
      v = xf[(gb + j) * HDIM + (c - 64)] - xf[gi * HDIM + (c - 64)];
    }
    eL[i] = (bf16)v;
  }
  __syncthreads();
  int lane = tid & 31, wv = tid >> 5;  // wave owns col-tile wv
  {
    // GEMM1: K=128. Weight fragments for this wave's col-tile are loop-invariant.
    v16bf bw1[4];
#pragma unroll
    for (int kb = 0; kb < 4; ++kb)
      bw1[kb] = load_b_frag(w1L + kb * 32 * 64 + (wv << 4), 64, lane);
    for (int rt = 0; rt < 6; ++rt) {
      v8f acc = {};
#pragma unroll
      for (int kb = 0; kb < 4; ++kb) {
        v16bf a = load_a_frag(eL + rt * 16 * 128 + kb * 32, 128, lane);
        acc = wmma_bf(a, bw1[kb], acc);
      }
#pragma unroll
      for (int r = 0; r < 8; ++r) {
        int m = rt * 16 + r + ((lane >> 4) << 3);
        int n = (wv << 4) + (lane & 15);
        h1L[m * 64 + n] = (bf16)eluf(acc[r] + b1L[n]);
      }
    }
  }
  __syncthreads();
  {
    // GEMM2: K=64, then max over k per node
    v16bf bw2[2];
#pragma unroll
    for (int kb = 0; kb < 2; ++kb)
      bw2[kb] = load_b_frag(w2L + kb * 32 * 64 + (wv << 4), 64, lane);
    for (int rt = 0; rt < 6; ++rt) {
      v8f acc = {};
#pragma unroll
      for (int kb = 0; kb < 2; ++kb) {
        v16bf a = load_a_frag(h1L + rt * 16 * 64 + kb * 32, 64, lane);
        acc = wmma_bf(a, bw2[kb], acc);
      }
#pragma unroll
      for (int r = 0; r < 8; ++r) {
        int m = rt * 16 + r + ((lane >> 4) << 3);
        int n = (wv << 4) + (lane & 15);
        float h = eluf(acc[r] + b2L[n]);
        atomicMaxF(&outL[(m / KNN) * 64 + n], h);
      }
    }
  }
  __syncthreads();
  for (int i = tid; i < 256; i += 128) feats[node0 * HDIM + i] = outL[i];
}

// ---- residual + LayerNorm; refresh bf16 copy and squared norms ----
__global__ void ln_kernel(const float* __restrict__ feats, const float* __restrict__ g,
                          const float* __restrict__ bb, float* __restrict__ xf,
                          bf16* __restrict__ xbf, float* __restrict__ sq) {
  int n = blockIdx.x * blockDim.x + threadIdx.x;
  if (n >= NTOT) return;
  float r[64]; float mu = 0.f;
#pragma unroll
  for (int c = 0; c < 64; ++c) { r[c] = feats[n * 64 + c] + xf[n * 64 + c]; mu += r[c]; }
  mu *= (1.f / 64.f);
  float var = 0.f;
#pragma unroll
  for (int c = 0; c < 64; ++c) { float d = r[c] - mu; var += d * d; }
  var *= (1.f / 64.f);
  float inv = rsqrtf(var + EPSV);
  float s = 0.f;
#pragma unroll
  for (int c = 0; c < 64; ++c) {
    float y = (r[c] - mu) * inv * g[c] + bb[c];
    xf[n * 64 + c] = y;
    xbf[n * 64 + c] = (bf16)y;
    s += y * y;
  }
  sq[n] = s;
}

// ---- head MLP: 64 -> 64(elu) -> 32(elu) -> 8 ----
__global__ void head_kernel(const float* __restrict__ xf, const float* w1,
                            const float* b1, const float* w2, const float* b2,
                            const float* w3, const float* b3, float* __restrict__ out) {
  __shared__ float sx[32 * 64], h1[32 * 64], h2[32 * 32];
  __shared__ float sw1[64 * 64], sw2[64 * 32], sw3[32 * 8], sb1[64], sb2[32], sb3[8];
  int tid = threadIdx.x;
  for (int i = tid; i < 64 * 64; i += 256) sw1[i] = w1[i];
  for (int i = tid; i < 64 * 32; i += 256) sw2[i] = w2[i];
  for (int i = tid; i < 32 * 8; i += 256) sw3[i] = w3[i];
  if (tid < 64) sb1[tid] = b1[tid];
  if (tid < 32) sb2[tid] = b2[tid];
  if (tid < 8) sb3[tid] = b3[tid];
  int n0 = blockIdx.x * 32;
  for (int i = tid; i < 32 * 64; i += 256) sx[i] = xf[n0 * 64 + i];
  __syncthreads();
  int ln = tid >> 3, g = tid & 7;
#pragma unroll
  for (int q = 0; q < 8; ++q) {
    int c = g * 8 + q; float a = sb1[c];
    for (int k = 0; k < 64; ++k) a += sx[ln * 64 + k] * sw1[k * 64 + c];
    h1[ln * 64 + c] = eluf(a);
  }
  __syncthreads();
#pragma unroll
  for (int q = 0; q < 4; ++q) {
    int c = g * 4 + q; float a = sb2[c];
    for (int k = 0; k < 64; ++k) a += h1[ln * 64 + k] * sw2[k * 32 + c];
    h2[ln * 32 + c] = eluf(a);
  }
  __syncthreads();
  {
    int c = g; float a = sb3[c];
    for (int k = 0; k < 32; ++k) a += h2[ln * 32 + k] * sw3[k * 8 + c];
    out[(n0 + ln) * 8 + c] = a;
  }
}

extern "C" void kernel_launch(void* const* d_in, const int* in_sizes, int n_in,
                              void* d_out, int out_size, void* d_ws, size_t ws_size,
                              hipStream_t stream) {
  // Input order (jax pytree: top-level dict order, nested dicts key-sorted):
  // 0 x_lc, 1 batch_lc,
  // conv{1,2,3}: b1,b2,w1,w2  -> 2..13
  // enc: b1,b2,b3,w1,w2,w3    -> 14..19
  // norm{1,2,3}: b,g          -> 20..25
  // out: b1,b2,b3,w1,w2,w3    -> 26..31
  const float* x_lc = (const float*)d_in[0];
  const float *cb1[3], *cb2[3], *cw1[3], *cw2[3];
  for (int l = 0; l < 3; ++l) {
    int b = 2 + 4 * l;
    cb1[l] = (const float*)d_in[b + 0];
    cb2[l] = (const float*)d_in[b + 1];
    cw1[l] = (const float*)d_in[b + 2];
    cw2[l] = (const float*)d_in[b + 3];
  }
  const float* eb1 = (const float*)d_in[14];
  const float* eb2 = (const float*)d_in[15];
  const float* eb3 = (const float*)d_in[16];
  const float* ew1 = (const float*)d_in[17];
  const float* ew2 = (const float*)d_in[18];
  const float* ew3 = (const float*)d_in[19];
  const float *nb[3], *ng[3];
  for (int l = 0; l < 3; ++l) {
    nb[l] = (const float*)d_in[20 + 2 * l];
    ng[l] = (const float*)d_in[21 + 2 * l];
  }
  const float* ob1 = (const float*)d_in[26];
  const float* ob2 = (const float*)d_in[27];
  const float* ob3 = (const float*)d_in[28];
  const float* ow1 = (const float*)d_in[29];
  const float* ow2 = (const float*)d_in[30];
  const float* ow3 = (const float*)d_in[31];

  char* ws = (char*)d_ws;                    // ~24.3 MB total (fits in L2)
  float* xf    = (float*)(ws + 0);           // [N,64] fp32
  float* feats = (float*)(ws + 8388608);     // [N,64] fp32
  bf16* xbf    = (bf16*)(ws + 16777216);     // [N,64] bf16
  float* sq    = (float*)(ws + 20971520);    // [N]
  int* idxb    = (int*)(ws + 21102592);      // [N,24]
  bf16* wbf    = (bf16*)(ws + 24248320);     // conv weights bf16, 3*12288

  cvt_w_kernel<<<144, 256, 0, stream>>>(cw1[0], cw2[0], cw1[1], cw2[1], cw1[2], cw2[2], wbf);
  enc_kernel<<<NTOT / 32, 256, 0, stream>>>(x_lc, ew1, eb1, ew2, eb2, ew3, eb3, xf);
  prep_kernel<<<NTOT / 256, 256, 0, stream>>>(xf, xbf, sq);
  for (int l = 0; l < 3; ++l) {
    knn_kernel<<<BGR * (NPG / 16), 128, 0, stream>>>(xbf, sq, idxb);
    edge_kernel<<<NTOT / 4, 128, 0, stream>>>(xf, wbf + (size_t)l * 12288, cb1[l], cb2[l],
                                              idxb, feats);
    ln_kernel<<<NTOT / 256, 256, 0, stream>>>(feats, ng[l], nb[l], xf, xbf, sq);
  }
  head_kernel<<<NTOT / 32, 256, 0, stream>>>(xf, ow1, ob1, ow2, ob2, ow3, ob3, (float*)d_out);
}
